// GRU_Att_Layers_30554397344422
// MI455X (gfx1250) — compile-verified, matrix operand order checked by reading, not run on hardware
//
#include <hip/hip_runtime.h>

// ---------------------------------------------------------------------------
// GRU + attention scan for MI455X (gfx1250), bf16 WMMA, persistent scan kernel
// B=64, T=512, H=768, L=2
// ---------------------------------------------------------------------------

#define BB 64
#define TT 512
#define HH 768
#define LL 2
#define H3 (3 * HH)
#define NWG 48          // H / 16 feature slices
#define KT (HH / 32)    // 24 k-steps of K=32 per GEMM

typedef __bf16 v16bf __attribute__((ext_vector_type(16)));
typedef float  v8f   __attribute__((ext_vector_type(8)));

union Frag { uint4 u[2]; v16bf v; };

__device__ __forceinline__ float sigmoidf_(float x) {
    return 1.0f / (1.0f + __expf(-x));
}

__device__ __forceinline__ v8f wmma_bf16(v16bf a, v16bf b, v8f c) {
    return __builtin_amdgcn_wmma_f32_16x16x32_bf16(
        false, a, false, b, (short)0, c, false, false);
}

// A-style 16x32 bf16 fragment load (also used for B as transposed weights):
// lane<16 : row = row0+lane,    k = kt*32 + {0..7, 16..23}
// lane>=16: row = row0+lane-16, k = kt*32 + {8..15, 24..31}
__device__ __forceinline__ v16bf load_frag(const __bf16* __restrict__ base,
                                           int pitch, int row0, int kt, int lane) {
    int r  = row0 + (lane & 15);
    int k0 = kt * 32 + ((lane & 16) >> 1);   // 0 or 8
    const __bf16* p = base + (size_t)r * pitch + k0;
    Frag f;
    f.u[0] = *(const uint4*)(p);
    f.u[1] = *(const uint4*)(p + 16);
    return f.v;
}

// Monotonic-counter grid barrier (48 resident WGs).
__device__ __forceinline__ void gridbar(int* cnt, int target) {
    __threadfence();
    __syncthreads();
    if (threadIdx.x == 0) {
        __hip_atomic_fetch_add(cnt, 1, __ATOMIC_ACQ_REL, __HIP_MEMORY_SCOPE_AGENT);
        while (__hip_atomic_load(cnt, __ATOMIC_ACQUIRE, __HIP_MEMORY_SCOPE_AGENT) < target) {
            __builtin_amdgcn_s_sleep(1);
        }
    }
    __syncthreads();
    __threadfence();
}

// ------------------------- conversion kernels ------------------------------

__global__ void cvt_f32_bf16(const float* __restrict__ src,
                             __bf16* __restrict__ dst, size_t n) {
    size_t i = (size_t)blockIdx.x * blockDim.x + threadIdx.x;
    size_t st = (size_t)gridDim.x * blockDim.x;
    for (; i < n; i += st) dst[i] = (__bf16)src[i];
}

// Wa: [rows][2H] -> Wax [rows][H], Wah [rows][H]
__global__ void cvt_wa(const float* __restrict__ Wa,
                       __bf16* __restrict__ Wax, __bf16* __restrict__ Wah,
                       size_t rows) {
    size_t n = rows * HH;
    size_t i = (size_t)blockIdx.x * blockDim.x + threadIdx.x;
    size_t st = (size_t)gridDim.x * blockDim.x;
    for (; i < n; i += st) {
        size_t r = i / HH, k = i % HH;
        Wax[i] = (__bf16)Wa[r * 2 * HH + k];
        Wah[i] = (__bf16)Wa[r * 2 * HH + HH + k];
    }
}

__global__ void init_layer(const float* __restrict__ hidden_l,
                           __bf16* __restrict__ hbuf0, int* __restrict__ bar_cnt) {
    int i = blockIdx.x * blockDim.x + threadIdx.x;
    if (i < BB * HH) hbuf0[i] = (__bf16)hidden_l[i];
    if (i == 0) *bar_cnt = 0;
}

// --------------------- preA = x @ Wax^T (big parallel GEMM) ----------------

__global__ void __launch_bounds__(128)
prea_kernel(const __bf16* __restrict__ xin,   // [B*T, H]
            const __bf16* __restrict__ Wax,   // [H, H]
            __bf16* __restrict__ preA) {      // [B*T, H]
    int lane = threadIdx.x & 31;
    int wave = threadIdx.x >> 5;
    int mt = blockIdx.x * 4 + wave;           // 0..2047
    int nt = blockIdx.y;                      // 0..47
    int mbase = mt * 16, fbase = nt * 16;
    v8f acc0 = {}, acc1 = {};
#pragma unroll 4
    for (int kt = 0; kt < KT; kt += 2) {      // two independent WMMA chains
        v16bf a0 = load_frag(xin, HH, mbase, kt, lane);
        v16bf b0 = load_frag(Wax, HH, fbase, kt, lane);
        v16bf a1 = load_frag(xin, HH, mbase, kt + 1, lane);
        v16bf b1 = load_frag(Wax, HH, fbase, kt + 1, lane);
        acc0 = wmma_bf16(a0, b0, acc0);
        acc1 = wmma_bf16(a1, b1, acc1);
    }
    v8f acc = acc0 + acc1;
    int col = lane & 15, rowoff = (lane >> 4) << 3;
    int f = fbase + col;
#pragma unroll
    for (int v = 0; v < 8; ++v) {
        int m = mbase + rowoff + v;
        preA[(size_t)m * HH + f] = (__bf16)acc[v];
    }
}

// ------------------------- persistent scan kernel --------------------------

__global__ void __launch_bounds__(128)
scan_kernel(const __bf16* __restrict__ xin,   // [B,T,H] layer input
            __bf16* __restrict__ yout,        // [B,T,H] layer output
            const __bf16* __restrict__ preA,  // [B,T,H]
            const __bf16* __restrict__ WiBf,  // [3H,H] this layer
            const __bf16* __restrict__ WhBf,  // [3H,H]
            const __bf16* __restrict__ WahBf, // [H,H]
            const float* __restrict__ bi,     // [3H]
            const float* __restrict__ bh,     // [3H]
            const float* __restrict__ ba,     // [H]
            const int* __restrict__ batch_idx,// [B,H]
            __bf16* __restrict__ hbuf0,
            __bf16* __restrict__ hbuf1,
            __bf16* __restrict__ xsbuf,       // [B,H] gated input, per step
            int* __restrict__ bar_cnt,
            float* __restrict__ out_nh,       // next_hidden[l] : [B,H] fp32
            float* __restrict__ out_final) {  // d_out[0:B*H] for last layer
    extern __shared__ char smem[];
    __bf16* sWi = (__bf16*)smem;          // [48][768]  (gate-major, 16 cols/gate)
    __bf16* sWh = sWi + 48 * HH;          // [48][768]
    __bf16* sWa = sWh + 48 * HH;          // [16][768]

    const int wg   = blockIdx.x;          // feature slice 0..47
    const int tid  = threadIdx.x;
    const int wave = tid >> 5;            // m-tile 0..3 (16 batch rows each)
    const int lane = tid & 31;
    const int fbase = wg * 16;
    const int mbase = wave * 16;

    // ---- stage weights into LDS (once) ----
    for (int r = tid; r < 112; r += 128) {
        const __bf16* src;
        __bf16* dst;
        if (r < 48) {
            int g = r / 16, c = r % 16;
            src = WiBf + (size_t)(g * HH + fbase + c) * HH;
            dst = sWi + r * HH;
        } else if (r < 96) {
            int rr = r - 48;
            int g = rr / 16, c = rr % 16;
            src = WhBf + (size_t)(g * HH + fbase + c) * HH;
            dst = sWh + rr * HH;
        } else {
            int c = r - 96;
            src = WahBf + (size_t)(fbase + c) * HH;
            dst = sWa + c * HH;
        }
        const uint4* s4 = (const uint4*)src;
        uint4* d4 = (uint4*)dst;
        for (int i = 0; i < HH / 8; ++i) d4[i] = s4[i];
    }
    __syncthreads();

    // ---- per-lane constants ----
    const int col = lane & 15;
    const int f = fbase + col;
    const int rowoff = (lane >> 4) << 3;
    float c_ba = ba[f];
    float c_bi0 = bi[0 * HH + f], c_bi1 = bi[1 * HH + f], c_bi2 = bi[2 * HH + f];
    float c_bh0 = bh[0 * HH + f], c_bh1 = bh[1 * HH + f], c_bh2 = bh[2 * HH + f];
    int brow[8], bidx[8];
#pragma unroll
    for (int v = 0; v < 8; ++v) {
        brow[v] = mbase + rowoff + v;
        bidx[v] = batch_idx[brow[v] * HH + f];
    }

    int bar = 0;
    for (int t = 0; t < TT; ++t) {
        const __bf16* hcur = (t & 1) ? hbuf1 : hbuf0;
        __bf16* hnext      = (t & 1) ? hbuf0 : hbuf1;

        // ---- Stage A: attention gate slice, xs = x * sigmoid(preA + h@Wah^T + ba)
        v8f accA0 = {}, accA1 = {};       // two independent WMMA chains
#pragma unroll 4
        for (int kt = 0; kt < KT; kt += 2) {
            v16bf ah0 = load_frag(hcur, HH, mbase, kt, lane);
            v16bf bw0 = load_frag(sWa, HH, 0, kt, lane);
            v16bf ah1 = load_frag(hcur, HH, mbase, kt + 1, lane);
            v16bf bw1 = load_frag(sWa, HH, 0, kt + 1, lane);
            accA0 = wmma_bf16(ah0, bw0, accA0);
            accA1 = wmma_bf16(ah1, bw1, accA1);
        }
        v8f accA = accA0 + accA1;
#pragma unroll
        for (int v = 0; v < 8; ++v) {
            int b = brow[v];
            size_t xoff = ((size_t)b * TT + t) * HH + f;
            float a = sigmoidf_(accA[v] + (float)preA[xoff] + c_ba);
            xsbuf[b * HH + f] = (__bf16)((float)xin[xoff] * a);
            // prefetch next timestep's activations while this step's math runs
            if (t + 1 < TT) {
                __builtin_prefetch(&preA[xoff + HH], 0, 1);
                __builtin_prefetch(&xin[xoff + HH], 0, 1);
            }
        }
        gridbar(bar_cnt, (++bar) * NWG);

        // ---- Stage B: gi = xs@Wi^T+bi, gh = h@Wh^T+bh, GRU combine
        v8f aIr = {}, aIz = {}, aIn = {}, aHr = {}, aHz = {}, aHn = {};
#pragma unroll 2
        for (int kt = 0; kt < KT; ++kt) {
            v16bf axs = load_frag(xsbuf, HH, mbase, kt, lane);
            v16bf ah  = load_frag(hcur,  HH, mbase, kt, lane);
            aIr = wmma_bf16(axs, load_frag(sWi, HH, 0,  kt, lane), aIr);
            aHr = wmma_bf16(ah,  load_frag(sWh, HH, 0,  kt, lane), aHr);
            aIz = wmma_bf16(axs, load_frag(sWi, HH, 16, kt, lane), aIz);
            aHz = wmma_bf16(ah,  load_frag(sWh, HH, 16, kt, lane), aHz);
            aIn = wmma_bf16(axs, load_frag(sWi, HH, 32, kt, lane), aIn);
            aHn = wmma_bf16(ah,  load_frag(sWh, HH, 32, kt, lane), aHn);
        }
#pragma unroll
        for (int v = 0; v < 8; ++v) {
            int b = brow[v];
            float hold = (float)hcur[b * HH + f];
            float r = sigmoidf_(aIr[v] + c_bi0 + aHr[v] + c_bh0);
            float z = sigmoidf_(aIz[v] + c_bi1 + aHz[v] + c_bh1);
            float n = tanhf(aIn[v] + c_bi2 + r * (aHn[v] + c_bh2));
            float hnew = (1.0f - z) * n + z * hold;
            __bf16 hb = (__bf16)hnew;
            hnext[b * HH + f] = hb;
            yout[((size_t)b * TT + t) * HH + f] = hb;
            if (bidx[v] == t) {
                out_nh[b * HH + f] = hnew;
                if (out_final) out_final[b * HH + f] = hnew;
            }
        }
        gridbar(bar_cnt, (++bar) * NWG);
    }
}

// ------------------------------- host side ---------------------------------

extern "C" void kernel_launch(void* const* d_in, const int* in_sizes, int n_in,
                              void* d_out, int out_size, void* d_ws, size_t ws_size,
                              hipStream_t stream) {
    const float* inputs    = (const float*)d_in[0];  // [B,T,H]
    const float* hidden    = (const float*)d_in[1];  // [L,B,H]
    const int*   batch_idx = (const int*)d_in[2];    // [B,1,H]
    const float* Wi        = (const float*)d_in[3];  // [L,3H,H]
    const float* Wh        = (const float*)d_in[4];  // [L,3H,H]
    const float* bi        = (const float*)d_in[5];  // [L,3H]
    const float* bh        = (const float*)d_in[6];  // [L,3H]
    const float* Wa        = (const float*)d_in[7];  // [L,H,2H]
    const float* ba        = (const float*)d_in[8];  // [L,H]
    float* out = (float*)d_out;                      // [B,H] ++ [L,B,H]

    // workspace carve
    char* ws = (char*)d_ws;
    auto carve = [&](size_t bytes) {
        char* p = ws;
        ws += (bytes + 255) & ~(size_t)255;
        return p;
    };
    const size_t nWi = (size_t)LL * H3 * HH;
    const size_t nWa = (size_t)LL * HH * HH;
    const size_t nX  = (size_t)BB * TT * HH;

    __bf16* WiBf  = (__bf16*)carve(nWi * 2);
    __bf16* WhBf  = (__bf16*)carve(nWi * 2);
    __bf16* WaxBf = (__bf16*)carve(nWa * 2);
    __bf16* WahBf = (__bf16*)carve(nWa * 2);
    __bf16* xb0   = (__bf16*)carve(nX * 2);
    __bf16* xb1   = (__bf16*)carve(nX * 2);
    __bf16* preA  = (__bf16*)carve(nX * 2);
    __bf16* hb0   = (__bf16*)carve((size_t)BB * HH * 2);
    __bf16* hb1   = (__bf16*)carve((size_t)BB * HH * 2);
    __bf16* xsb   = (__bf16*)carve((size_t)BB * HH * 2);
    int*    barc  = (int*)carve(256);

    // conversions
    cvt_f32_bf16<<<2048, 256, 0, stream>>>(Wi, WiBf, nWi);
    cvt_f32_bf16<<<2048, 256, 0, stream>>>(Wh, WhBf, nWi);
    cvt_wa<<<2048, 256, 0, stream>>>(Wa, WaxBf, WahBf, (size_t)LL * HH);
    cvt_f32_bf16<<<4096, 256, 0, stream>>>(inputs, xb0, nX);

    const size_t smem = (size_t)(48 * HH + 48 * HH + 16 * HH) * sizeof(__bf16); // 168 KB

    const __bf16* xin = xb0;
    __bf16* yout = xb1;
    for (int l = 0; l < LL; ++l) {
        init_layer<<<(BB * HH + 255) / 256, 256, 0, stream>>>(hidden + (size_t)l * BB * HH, hb0, barc);
        prea_kernel<<<dim3(512, 48), 128, 0, stream>>>(xin, WaxBf + (size_t)l * HH * HH, preA);
        scan_kernel<<<NWG, 128, smem, stream>>>(
            xin, yout, preA,
            WiBf + (size_t)l * H3 * HH, WhBf + (size_t)l * H3 * HH,
            WahBf + (size_t)l * HH * HH,
            bi + (size_t)l * H3, bh + (size_t)l * H3, ba + (size_t)l * HH,
            batch_idx, hb0, hb1, xsb, barc,
            out + (size_t)BB * HH + (size_t)l * BB * HH,
            (l == LL - 1) ? out : (float*)nullptr);
        // swap buffers: next layer consumes this layer's output
        xin = yout;
        yout = (l == 0) ? xb0 : xb1;
    }
    (void)in_sizes; (void)n_in; (void)out_size; (void)ws_size;
}
